// CorrLayer_4827543241305
// MI455X (gfx1250) — compile-verified, matrix operand order checked by reading, not run on hardware
//
#include <hip/hip_runtime.h>
#include <hip/hip_bf16.h>
#include <math.h>

typedef __attribute__((ext_vector_type(16))) __bf16 v16bf;
typedef __attribute__((ext_vector_type(2)))  __bf16 bf16x2;
typedef __attribute__((ext_vector_type(8)))  float  v8f;
typedef __attribute__((ext_vector_type(4)))  float  float4v;

#define L_SEQ  2048
#define NB     4
#define NFREQ  1025
#define FPAD   1040   // NFREQ padded to multiple of 16
#define KTOP   15     // int(2*ln(2048)) = 15

// Native f32 -> bf16 conversion (RNE); lowers to the hardware convert on gfx1250.
__device__ __forceinline__ unsigned short f2bf(float x) {
  __bf16 h = (__bf16)x;
  return __builtin_bit_cast(unsigned short, h);
}

// Pack two f32 -> one dword of two bf16; matches v_cvt_pk_bf16_f32.
__device__ __forceinline__ unsigned int f2bf2(float lo, float hi) {
  bf16x2 h = { (__bf16)lo, (__bf16)hi };
  return __builtin_bit_cast(unsigned int, h);
}

__device__ __forceinline__ v8f zero8() {
  v8f z = {0.f,0.f,0.f,0.f,0.f,0.f,0.f,0.f};
  return z;
}

// Load one 16x32 bf16 WMMA operand fragment from an LDS tile stored
// row-major with 32 bf16 per row. Lane pattern per CDNA5 ISA 7.12.2:
// lanes 0-15 hold K=0..7 and 16..23, lanes 16-31 hold K=8..15 and 24..31.
__device__ __forceinline__ v16bf load_frag(const unsigned short* rowp, int kbase) {
  union { unsigned int u[8]; v16bf v; } r;
#pragma unroll
  for (int p = 0; p < 4; ++p)
    r.u[p] = *(const unsigned int*)(rowp + kbase + 2 * p);
#pragma unroll
  for (int p = 0; p < 4; ++p)
    r.u[4 + p] = *(const unsigned int*)(rowp + 16 + kbase + 2 * p);
  return r.v;
}

// Generic WMMA GEMM: Out(MxN) = A(MxK) * B(KxN) (+bias), fp32 in/out,
// bf16 v_wmma_f32_16x16x32_bf16 with fp32 accumulation.
// A_MODE:   0 = row-major lda; 1 = batch-transposed A[b][col][t], b=row>>11, t=row&2047
// OUT_MODE: 0 = row-major ldo; 1 = batch-transposed Out[b][col][t]
// B_BF16:   0 = B is fp32 row-major;  1 = B is bf16 row-major
template<int A_MODE, int OUT_MODE, int B_BF16>
__global__ __launch_bounds__(256) void wmma_gemm_kernel(
    const float* __restrict__ A, const void* __restrict__ Bm,
    const float* __restrict__ bias, float* __restrict__ Out,
    int M, int N, int K, int lda, int ldb, int ldo)
{
  __shared__ unsigned short ldsA[128 * 32];  // [m][k]
  __shared__ unsigned short ldsB[128 * 32];  // [n][k] (transposed on fill)

  const int tid  = threadIdx.x;
  const int lane = tid & 31;
  const int wave = tid >> 5;
  const int waveM = (wave & 1) * 64;   // 2 waves along M
  const int waveN = (wave >> 1) * 32;  // 4 waves along N
  const int blockM = blockIdx.y * 128;
  const int blockN = blockIdx.x * 128;
  const bool fullN = (blockN + 128 <= N);

  v8f acc[4][2];
#pragma unroll
  for (int m = 0; m < 4; ++m)
#pragma unroll
    for (int n = 0; n < 2; ++n) acc[m][n] = zero8();

  const float*          Bf = (const float*)Bm;
  const unsigned short* Bh = (const unsigned short*)Bm;

  // Software-pipeline staging registers (fetch k+1 while WMMAs consume k).
  float4v ar[4];
  float4v brf[4];
  unsigned long long brh[4];

  auto fetchA = [&](int k0) {
    if constexpr (A_MODE == 0) {
#pragma unroll
      for (int it = 0; it < 4; ++it) {
        int i4 = tid + it * 256;             // 1024 float4 in a 128x32 tile
        int row = i4 >> 3, col = (i4 & 7) * 4;   // coalesced along K
        ar[it] = *(const float4v*)(A + (long)(blockM + row) * lda + (k0 + col));
      }
    } else {
      const int b = blockM >> 11, t0 = blockM & 2047;
      const float* Ab = A + (((long)b * K) << 11) + t0;
#pragma unroll
      for (int it = 0; it < 4; ++it) {
        int i4 = tid + it * 256;
        int col = i4 >> 5, rowb = (i4 & 31) * 4; // coalesced along t (rows)
        ar[it] = *(const float4v*)(Ab + ((long)(k0 + col) << 11) + rowb);
      }
    }
  };

  auto fetchB = [&](int k0) {
#pragma unroll
    for (int it = 0; it < 4; ++it) {
      int i4 = tid + it * 256;
      int k = i4 >> 5, n4 = (i4 & 31) * 4;       // coalesced along N
      int gn = blockN + n4;
      if constexpr (B_BF16 == 1) {
        if (fullN || gn + 3 < N) {
          brh[it] = *(const unsigned long long*)(Bh + (long)(k0 + k) * ldb + gn);
        } else {
          unsigned long long r = 0;
#pragma unroll
          for (int j = 0; j < 4; ++j)
            if (gn + j < N)
              r |= (unsigned long long)Bh[(long)(k0 + k) * ldb + gn + j] << (16 * j);
          brh[it] = r;
        }
      } else {
        if (fullN || gn + 3 < N) {
          brf[it] = *(const float4v*)(Bf + (long)(k0 + k) * ldb + gn);
        } else {
          float4v v = {0.f, 0.f, 0.f, 0.f};
#pragma unroll
          for (int j = 0; j < 4; ++j)
            if (gn + j < N) v[j] = Bf[(long)(k0 + k) * ldb + gn + j];
          brf[it] = v;
        }
      }
    }
  };

  auto stage = [&]() {
    if constexpr (A_MODE == 0) {
#pragma unroll
      for (int it = 0; it < 4; ++it) {
        int i4 = tid + it * 256;
        int row = i4 >> 3, col = (i4 & 7) * 4;
        unsigned int p0 = f2bf2(ar[it].x, ar[it].y);   // v_cvt_pk_bf16_f32
        unsigned int p1 = f2bf2(ar[it].z, ar[it].w);
        *(unsigned int*)&ldsA[row * 32 + col]     = p0;  // merges to ds_store_b64
        *(unsigned int*)&ldsA[row * 32 + col + 2] = p1;
      }
    } else {
#pragma unroll
      for (int it = 0; it < 4; ++it) {
        int i4 = tid + it * 256;
        int col = i4 >> 5, rowb = (i4 & 31) * 4;
#pragma unroll
        for (int j = 0; j < 4; ++j)
          ldsA[(rowb + j) * 32 + col] = f2bf(ar[it][j]);
      }
    }
#pragma unroll
    for (int it = 0; it < 4; ++it) {
      int i4 = tid + it * 256;
      int k = i4 >> 5, n4 = (i4 & 31) * 4;
#pragma unroll
      for (int j = 0; j < 4; ++j) {
        unsigned short hv;
        if constexpr (B_BF16 == 1) hv = (unsigned short)(brh[it] >> (16 * j));
        else                       hv = f2bf(brf[it][j]);
        ldsB[(n4 + j) * 32 + k] = hv;
      }
    }
  };

  fetchA(0);
  fetchB(0);

  for (int k0 = 0; k0 < K; k0 += 32) {
    stage();
    __syncthreads();
    if (k0 + 32 < K) { fetchA(k0 + 32); fetchB(k0 + 32); }  // overlap with WMMA

    const int kbase = (lane < 16) ? 0 : 8;
    v16bf afrag[4], bfrag[2];
#pragma unroll
    for (int m = 0; m < 4; ++m)
      afrag[m] = load_frag(&ldsA[(waveM + m * 16 + (lane & 15)) * 32], kbase);
#pragma unroll
    for (int n = 0; n < 2; ++n)
      bfrag[n] = load_frag(&ldsB[(waveN + n * 16 + (lane & 15)) * 32], kbase);

#pragma unroll
    for (int m = 0; m < 4; ++m)
#pragma unroll
      for (int n = 0; n < 2; ++n)
        acc[m][n] = __builtin_amdgcn_wmma_f32_16x16x32_bf16(
            false, afrag[m], false, bfrag[n], (short)0, acc[m][n], false, false);
    __syncthreads();
  }

  // Epilogue. D layout: VGPR r: lanes 0-15 M=r, N=lane; lanes 16-31 M=r+8.
  const int mhalf = (lane >> 4) * 8;
  const int ncol  = lane & 15;
#pragma unroll
  for (int m = 0; m < 4; ++m)
#pragma unroll
    for (int n = 0; n < 2; ++n)
#pragma unroll
      for (int r = 0; r < 8; ++r) {
        int grow = blockM + waveM + m * 16 + mhalf + r;
        int gcol = blockN + waveN + n * 16 + ncol;
        if (grow < M && gcol < N) {
          float v = acc[m][n][r];
          if (bias) v += bias[gcol];
          if constexpr (OUT_MODE == 0) {
            Out[(long)grow * ldo + gcol] = v;
          } else {
            int b = grow >> 11, t = grow & 2047;
            Out[(((long)b * N) << 11) + ((long)gcol << 11) + t] = v;
          }
        }
      }
}

// DFT bases: cos(2*pi*f*t/L), sin(2*pi*f*t/L), bf16, exact mod-L angle reduction.
__global__ void gen_basis_kernel(unsigned short* __restrict__ cosb,
                                 unsigned short* __restrict__ sinb) {
  int idx = blockIdx.x * 256 + threadIdx.x;
  if (idx >= L_SEQ * FPAD) return;
  int t = idx / FPAD, f = idx % FPAD;
  unsigned short cv = 0, sv = 0;
  if (f < NFREQ) {
    int m = (f * t) & (L_SEQ - 1);
    float ang = (float)m * (6.283185307179586f / (float)L_SEQ);
    float s, c;
    __sincosf(ang, &s, &c);
    cv = f2bf(c); sv = f2bf(s);
  }
  cosb[idx] = cv; sinb[idx] = sv;
}

// SAC(b,f) = sum_c Aqc*Akc ; SBD(b,f) = sum_c Aqs*Aks  (channel reduction)
__global__ void reduce_spec_kernel(const float* __restrict__ AQC, const float* __restrict__ AQS,
                                   const float* __restrict__ AKC, const float* __restrict__ AKS,
                                   float* __restrict__ SAC, float* __restrict__ SBD) {
  int idx = blockIdx.x * 256 + threadIdx.x;
  if (idx >= NB * FPAD) return;
  int b = idx / FPAD, f = idx % FPAD;
  if (f >= NFREQ) return;
  float sac = 0.f, sbd = 0.f;
  long base = (long)b * 1024 * FPAD + f;
  for (int c = 0; c < 1024; ++c) {
    long o = base + (long)c * FPAD;
    sac += AQC[o] * AKC[o];
    sbd += AQS[o] * AKS[o];
  }
  SAC[idx] = sac; SBD[idx] = sbd;
}

// irfft synthesis of the channel-mean correlation (c = SAC - i*SBD):
// mean(b,t) = (1/(L*1024)) [SAC0 + 2*sum_f(SAC*cos + SBD*sin) + (-1)^t * SAC_Nyq]
__global__ void synth_mean_kernel(const float* __restrict__ SAC,
                                  const float* __restrict__ SBD,
                                  float* __restrict__ mean) {
  int idx = blockIdx.x * 256 + threadIdx.x;
  if (idx >= NB * L_SEQ) return;
  int b = idx >> 11, t = idx & 2047;
  const float* sa = SAC + b * FPAD;
  const float* sb = SBD + b * FPAD;
  float sum = sa[0] + ((t & 1) ? -sa[1024] : sa[1024]);
  for (int f = 1; f < 1024; ++f) {
    int m = (f * t) & 2047;
    float ang = (float)m * (6.283185307179586f / 2048.0f);
    float s, c;
    __sincosf(ang, &s, &c);
    sum += 2.0f * (sa[f] * c + sb[f] * s);
  }
  mean[idx] = sum * (1.0f / (2048.0f * 1024.0f));
}

// Per-batch top-15 over L, then softmax of the winning values.
__global__ __launch_bounds__(256) void topk_softmax_kernel(
    const float* __restrict__ mean, float* __restrict__ wgt, int* __restrict__ dly) {
  __shared__ float vals[2048];
  __shared__ float svv[256];
  __shared__ int   sii[256];
  __shared__ float tw[KTOP];
  __shared__ int   td[KTOP];
  int b = blockIdx.x, tid = threadIdx.x;
  for (int i = tid; i < 2048; i += 256) vals[i] = mean[b * 2048 + i];
  __syncthreads();
  for (int r = 0; r < KTOP; ++r) {
    float best = -3.0e38f; int bi = 0;
    for (int i = tid; i < 2048; i += 256) {
      float v = vals[i];
      if (v > best) { best = v; bi = i; }
    }
    svv[tid] = best; sii[tid] = bi;
    __syncthreads();
    for (int s = 128; s > 0; s >>= 1) {
      if (tid < s && svv[tid + s] > svv[tid]) { svv[tid] = svv[tid + s]; sii[tid] = sii[tid + s]; }
      __syncthreads();
    }
    if (tid == 0) { tw[r] = svv[0]; td[r] = sii[0]; vals[sii[0]] = -3.0e38f; }
    __syncthreads();
  }
  if (tid == 0) {
    float mx = tw[0], s = 0.f, e[KTOP];
    for (int r = 0; r < KTOP; ++r) { e[r] = __expf(tw[r] - mx); s += e[r]; }
    for (int r = 0; r < KTOP; ++r) { wgt[b * KTOP + r] = e[r] / s; dly[b * KTOP + r] = td[r]; }
  }
}

// agg[b][c][t] = sum_j w_j * v[b][c][(t + delay_j) mod L]   (coalesced in t)
__global__ void agg_kernel(const float* __restrict__ VT, const float* __restrict__ wgt,
                           const int* __restrict__ dly, float* __restrict__ AGG) {
  int b = blockIdx.z, c = blockIdx.y;
  int t = blockIdx.x * 256 + threadIdx.x;
  __shared__ float w[KTOP];
  __shared__ int   d[KTOP];
  if (threadIdx.x < KTOP) {
    w[threadIdx.x] = wgt[b * KTOP + threadIdx.x];
    d[threadIdx.x] = dly[b * KTOP + threadIdx.x];
  }
  __syncthreads();
  const float* v = VT + ((long)b * 1024 + c) * 2048;
  float acc = 0.f;
#pragma unroll
  for (int j = 0; j < KTOP; ++j) acc += w[j] * v[(t + d[j]) & 2047];
  AGG[((long)b * 1024 + c) * 2048 + t] = acc;
}

extern "C" void kernel_launch(void* const* d_in, const int* in_sizes, int n_in,
                              void* d_out, int out_size, void* d_ws, size_t ws_size,
                              hipStream_t stream) {
  (void)in_sizes; (void)n_in; (void)out_size; (void)ws_size;
  const float* Q  = (const float*)d_in[0];
  const float* Kk = (const float*)d_in[1];
  const float* V  = (const float*)d_in[2];
  const float* Wq = (const float*)d_in[3];
  const float* bq = (const float*)d_in[4];
  const float* Wk = (const float*)d_in[5];
  const float* bk = (const float*)d_in[6];
  const float* Wv = (const float*)d_in[7];
  const float* bv = (const float*)d_in[8];
  const float* Wo = (const float*)d_in[9];
  const float* bo = (const float*)d_in[10];

  char* ws = (char*)d_ws;
  size_t off = 0;
  const size_t sz_qkv  = (size_t)NB * 1024 * 2048 * sizeof(float);   // 33.55 MB
  const size_t sz_bas  = (size_t)L_SEQ * FPAD * sizeof(unsigned short);
  const size_t sz_spec = (size_t)NB * 1024 * FPAD * sizeof(float);   // 17.04 MB

  float* QT = (float*)(ws + off); off += sz_qkv;
  float* KT = (float*)(ws + off); off += sz_qkv;
  float* VT = (float*)(ws + off); off += sz_qkv;
  unsigned short* COSB = (unsigned short*)(ws + off); off += sz_bas;
  unsigned short* SINB = (unsigned short*)(ws + off); off += sz_bas;
  float* AQC = (float*)(ws + off); off += sz_spec;
  float* AQS = (float*)(ws + off); off += sz_spec;
  float* AKC = (float*)(ws + off); off += sz_spec;
  float* AKS = (float*)(ws + off); off += sz_spec;
  float* SAC = (float*)(ws + off); off += (size_t)NB * FPAD * sizeof(float);
  float* SBD = (float*)(ws + off); off += (size_t)NB * FPAD * sizeof(float);
  float* MEAN = (float*)(ws + off); off += (size_t)NB * L_SEQ * sizeof(float);
  float* WGT = (float*)(ws + off); off += 256;
  int*   DLY = (int*)(ws + off);   off += 256;
  float* AGG = AQC;  // spectra dead after synth; reuse AQC+AQS (34 MB) for agg

  dim3 blk(256);

  // 0) DFT bases
  gen_basis_kernel<<<(L_SEQ * FPAD + 255) / 256, blk, 0, stream>>>(COSB, SINB);

  // 1) Q/K/V projections -> channel-major (B, c, t)
  dim3 gProj(1024 / 128, 8192 / 128);
  wmma_gemm_kernel<0, 1, 0><<<gProj, blk, 0, stream>>>(Q,  Wq, bq, QT, 8192, 1024, 1024, 1024, 1024, 0);
  wmma_gemm_kernel<0, 1, 0><<<gProj, blk, 0, stream>>>(Kk, Wk, bk, KT, 8192, 1024, 1024, 1024, 1024, 0);
  wmma_gemm_kernel<0, 1, 0><<<gProj, blk, 0, stream>>>(V,  Wv, bv, VT, 8192, 1024, 1024, 1024, 1024, 0);

  // 2) rFFTs as DFT GEMMs: (4096 x 2048) x (2048 x 1025)
  dim3 gDft((NFREQ + 127) / 128, 4096 / 128);
  wmma_gemm_kernel<0, 0, 1><<<gDft, blk, 0, stream>>>(QT, COSB, nullptr, AQC, 4096, NFREQ, 2048, 2048, FPAD, FPAD);
  wmma_gemm_kernel<0, 0, 1><<<gDft, blk, 0, stream>>>(QT, SINB, nullptr, AQS, 4096, NFREQ, 2048, 2048, FPAD, FPAD);
  wmma_gemm_kernel<0, 0, 1><<<gDft, blk, 0, stream>>>(KT, COSB, nullptr, AKC, 4096, NFREQ, 2048, 2048, FPAD, FPAD);
  wmma_gemm_kernel<0, 0, 1><<<gDft, blk, 0, stream>>>(KT, SINB, nullptr, AKS, 4096, NFREQ, 2048, 2048, FPAD, FPAD);

  // 3) channel-mean spectrum, 4) irfft synthesis
  reduce_spec_kernel<<<(NB * FPAD + 255) / 256, blk, 0, stream>>>(AQC, AQS, AKC, AKS, SAC, SBD);
  synth_mean_kernel<<<(NB * L_SEQ + 255) / 256, blk, 0, stream>>>(SAC, SBD, MEAN);

  // 5) top-15 + softmax, 6) weighted circular gather of V
  topk_softmax_kernel<<<NB, blk, 0, stream>>>(MEAN, WGT, DLY);
  agg_kernel<<<dim3(L_SEQ / 256, 1024, NB), blk, 0, stream>>>(VT, WGT, DLY, AGG);

  // 7) output projection (A read batch-transposed) + bias -> d_out
  wmma_gemm_kernel<1, 0, 0><<<gProj, blk, 0, stream>>>(AGG, Wo, bo, (float*)d_out,
                                                       8192, 1024, 1024, 0, 1024, 1024);
}
